// AgentLSM_70841190580507
// MI455X (gfx1250) — matrix operand order repaired and checked by reading, not compiled, not run on hardware
//
#include <hip/hip_runtime.h>
#include <hip/hip_bf16.h>

// CDNA5 / gfx1250, wave32. Fused CPG-LIF kernel:
//   - Tensor Data Mover (tensor_load_to_lds) DMAs the per-wave batch tiles
//     (neighbor spikes / v0 / spike_history) into LDS, tracked by TENSORcnt.
//   - v_wmma_f32_16x16x32_f16 for the coupling GEMM and the per-tick
//     recurrent GEMM (base drive folded into the WMMA C operand).
//   - LIF update / spike / accumulate in f32 registers (C/D layout), spike
//     transpose D-layout -> A-layout via per-wave LDS staging.

typedef __attribute__((ext_vector_type(16))) _Float16     v16h;
typedef __attribute__((ext_vector_type(8)))  float        v8f;
typedef __attribute__((ext_vector_type(4)))  unsigned int u32x4;
typedef __attribute__((ext_vector_type(8)))  int          i32x8;
typedef __attribute__((ext_vector_type(4)))  int          i32x4;

#define WAVES_PER_BLOCK 4
#define TILE_ROWS       16
#define ROWS_PER_BLOCK  (WAVES_PER_BLOCK * TILE_ROWS)
#define NUM_E           25
#define FATIGUE_TH      8

// 1-D contiguous TDM copy: nelem 4-byte elements, global -> LDS.
// D# layout per CDNA5 ISA:
//   group0: [1:0]=count=1 | lds_addr | global_addr[31:0] |
//           global_addr[56:32] + type=2 in [127:126]
//   group1: data_size=2 (4B) in [17:16]; tensor_dim0=nelem; tensor_dim1=1;
//           tile_dim0=nelem; tile_dim1/2=0; tensor_dim0_stride=nelem
__device__ __forceinline__ void tdm_load_1d(const void* gptr, void* lptr,
                                            unsigned nelem) {
    unsigned long long ga = (unsigned long long)(uintptr_t)gptr;
    unsigned la = (unsigned)(uintptr_t)lptr;   // low 32 bits = LDS offset
    u32x4 g0 = {
        1u,                                            // count=1, user mode
        la,                                            // lds_addr (bytes)
        (unsigned)(ga & 0xFFFFFFFFu),                  // global_addr[31:0]
        (unsigned)((ga >> 32) & 0x1FFFFFFu) | (2u << 30) // [56:32] | type=2
    };
    i32x8 g1 = {
        (int)(2u << 16),                       // workgroup_mask=0, data_size=4B
        (int)((nelem & 0xFFFFu) << 16),        // [63:48]  = tensor_dim0[15:0]
        (int)((nelem >> 16) | (1u << 16)),     // [79:64]=dim0 hi, [95:80]=tensor_dim1=1
        (int)((nelem & 0xFFFFu) << 16),        // [111:96]=dim1 hi=0, [127:112]=tile_dim0
        0,                                     // tile_dim1=0, tile_dim2=0
        (int)nelem,                            // tensor_dim0_stride[31:0]
        0,                                     // stride0[47:32]=0, stride1 lo=0
        0                                      // stride1 hi=0
    };
    i32x4 z4 = {0, 0, 0, 0};
    i32x8 z8 = {0, 0, 0, 0, 0, 0, 0, 0};
    __builtin_amdgcn_tensor_load_to_lds(g0, g1, z4, z4, z8, 0);
}

__global__ __launch_bounds__(WAVES_PER_BLOCK * 32) void AgentLSM_cpg_kernel(
    const float* __restrict__ sh_g,    // spike_history   [B,32]
    const float* __restrict__ nbr_g,   // neighbor spikes [B,8,25]
    const float* __restrict__ M_g,     // mutual_inhibition [32,32]
    const float* __restrict__ W_g,     // coupling_W [32,25]
    const float* __restrict__ v0_g,    // v0 [B,32]
    const int*   __restrict__ pf_g,    // peak_fatigue [B]
    const int*   __restrict__ tf_g,    // trough_fatigue [B]
    const int*   __restrict__ ticks_g, // num_ticks (scalar)
    float* __restrict__ out_acc,       // [B,25]
    float* __restrict__ out_stay,      // [B]
    int Btot)
{
    __shared__ float    sN  [WAVES_PER_BLOCK][TILE_ROWS * 8 * NUM_E]; // nbr tile
    __shared__ float    sV  [WAVES_PER_BLOCK][TILE_ROWS * 32];        // v0 tile
    __shared__ float    sH  [WAVES_PER_BLOCK][TILE_ROWS * 32];        // history tile
    __shared__ _Float16 sS  [WAVES_PER_BLOCK][TILE_ROWS][32];         // spike transpose
    __shared__ float    sAcc[WAVES_PER_BLOCK][TILE_ROWS][32];         // out staging

    const int tid  = threadIdx.x;
    const int w    = tid >> 5;        // wave in block (wave32)
    const int lane = tid & 31;
    const int col  = lane & 15;       // N column / A-row-in-tile
    const int hi   = lane >> 4;       // upper-half-lane flag
    const int T    = ticks_g[0];

    int b0 = (blockIdx.x * WAVES_PER_BLOCK + w) * TILE_ROWS;
    if (b0 + TILE_ROWS > Btot) b0 = Btot - TILE_ROWS;   // clamp tail tile
    if (b0 < 0) b0 = 0;

    // -------- TDM: DMA the three per-wave batch tiles into LDS --------------
    tdm_load_1d(nbr_g + (size_t)b0 * 8 * NUM_E, &sN[w][0], TILE_ROWS * 8 * NUM_E);
    tdm_load_1d(v0_g  + (size_t)b0 * 32,        &sV[w][0], TILE_ROWS * 32);
    tdm_load_1d(sh_g  + (size_t)b0 * 32,        &sH[w][0], TILE_ROWS * 32);

    // -------- B-matrix registers (ISA B layout, loop-invariant) -------------
    // lane -> column N = col, e<8 -> K = hi*8 + e, e>=8 -> K = 16 + hi*8 + (e-8)
    v16h bM0, bM1, bW0, bW1;
    {
        const int kb = hi * 8;
        #pragma unroll
        for (int e = 0; e < 16; ++e) {
            int k = (e < 8) ? kb + e : 16 + kb + (e - 8);
            bM0[e] = (_Float16)(0.5f * M_g[(size_t)col * 32 + k]);
            bM1[e] = (_Float16)(0.5f * M_g[(size_t)(16 + col) * 32 + k]);
            float w0 = (k < NUM_E) ? 0.3f * W_g[(size_t)col       * NUM_E + k] : 0.f;
            float w1 = (k < NUM_E) ? 0.3f * W_g[(size_t)(16 + col) * NUM_E + k] : 0.f;
            bW0[e] = (_Float16)w0;
            bW1[e] = (_Float16)w1;
        }
    }

    // -------- fatigue drive as WMMA C operand (C/D layout) ------------------
    v8f c0, c1;
    #pragma unroll
    for (int j = 0; j < 8; ++j) {
        int row = b0 + j + 8 * hi;
        c0[j] = (tf_g[row] >= FATIGUE_TH) ? 2.0f : 0.0f;  // peak half <- trough fat.
        c1[j] = (pf_g[row] >= FATIGUE_TH) ? 2.0f : 0.0f;  // trough half <- peak fat.
    }

    // -------- wait for TDM tiles, then consume from LDS ---------------------
    __builtin_amdgcn_s_wait_tensorcnt(0);
    asm volatile("" ::: "memory");

    // coupling A (ISA A layout): lane row r=col, on-the-fly k-reduction
    v16h a;
    {
        const int nb = hi * 8;
        const float* r = &sN[w][col * (8 * NUM_E)];
        #pragma unroll
        for (int e = 0; e < 8; ++e) {
            int n0 = nb + e;
            int n1 = 16 + nb + e;
            float s0 = 0.f, s1 = 0.f;
            #pragma unroll
            for (int k = 0; k < 8; ++k) {
                s0 += r[k * NUM_E + n0];
                if (n1 < NUM_E) s1 += r[k * NUM_E + n1];
            }
            a[e]     = (_Float16)s0;
            a[8 + e] = (_Float16)s1;
        }
    }

    // base = drive + 0.3 * coupling  (two 16x16x32 f16 WMMAs)
    v8f base0 = __builtin_amdgcn_wmma_f32_16x16x32_f16(false, a, false, bW0,
                                                       (short)0, c0, false, false);
    v8f base1 = __builtin_amdgcn_wmma_f32_16x16x32_f16(false, a, false, bW1,
                                                       (short)0, c1, false, false);

    // membrane v (C/D layout) from LDS v0 tile + accumulators
    v8f vr0, vr1;
    v8f acc0 = {0.f,0.f,0.f,0.f,0.f,0.f,0.f,0.f};
    v8f acc1 = {0.f,0.f,0.f,0.f,0.f,0.f,0.f,0.f};
    #pragma unroll
    for (int j = 0; j < 8; ++j) {
        int r = j + 8 * hi;
        vr0[j] = sV[w][r * 32 + col];
        vr1[j] = sV[w][r * 32 + 16 + col];
    }

    // initial spikes (A layout) from LDS history tile
    {
        const int nb = hi * 8;
        const float* p = &sH[w][col * 32];
        #pragma unroll
        for (int e = 0; e < 8; ++e) {
            a[e]     = (_Float16)p[nb + e];
            a[8 + e] = (_Float16)p[16 + nb + e];
        }
    }

    // ------------------------------ tick loop -------------------------------
    for (int t = 0; t < T; ++t) {
        // x = base + 0.5 * (spikes @ M^T)   (C operand carries base)
        v8f x0 = __builtin_amdgcn_wmma_f32_16x16x32_f16(false, a, false, bM0,
                                                        (short)0, base0, false, false);
        v8f x1 = __builtin_amdgcn_wmma_f32_16x16x32_f16(false, a, false, bM1,
                                                        (short)0, base1, false, false);
        v8f s0, s1;
        #pragma unroll
        for (int j = 0; j < 8; ++j) {
            float v0n = vr0[j] + (x0[j] - vr0[j]) * (1.0f / 3.0f);
            float v1n = vr1[j] + (x1[j] - vr1[j]) * (1.0f / 3.0f);
            float sp0 = (v0n >= 1.0f) ? 1.0f : 0.0f;
            float sp1 = (v1n >= 1.0f) ? 1.0f : 0.0f;
            acc0[j] += sp0;  acc1[j] += sp1;
            vr0[j] = v0n * (1.0f - sp0);
            vr1[j] = v1n * (1.0f - sp1);
            s0[j] = sp0;  s1[j] = sp1;
        }
        // spike transpose: D layout -> LDS -> A layout for next tick
        #pragma unroll
        for (int j = 0; j < 8; ++j) {
            sS[w][j + 8 * hi][col]      = (_Float16)s0[j];
            sS[w][j + 8 * hi][16 + col] = (_Float16)s1[j];
        }
        __syncthreads();
        {
            const int nb = hi * 8;
            #pragma unroll
            for (int e = 0; e < 8; ++e) {
                a[e]     = sS[w][col][nb + e];
                a[8 + e] = sS[w][col][16 + nb + e];
            }
        }
        __syncthreads();
    }

    // ------------------------------ outputs ---------------------------------
    #pragma unroll
    for (int j = 0; j < 8; ++j) {
        sAcc[w][j + 8 * hi][col]      = acc0[j];
        sAcc[w][j + 8 * hi][16 + col] = acc1[j];
    }
    __syncthreads();

    for (int idx = lane; idx < TILE_ROWS * NUM_E; idx += 32) {
        int bl = idx / NUM_E;
        int m  = idx - bl * NUM_E;
        int row = b0 + bl;
        if (row < Btot) out_acc[(size_t)row * NUM_E + m] = sAcc[w][bl][m];
    }
    if (lane < TILE_ROWS) {
        int row = b0 + lane;
        if (row < Btot) {
            float s = 0.f;
            #pragma unroll
            for (int m = 16; m < 32; ++m) s += sAcc[w][lane][m];
            out_stay[row] = (s > 0.5f) ? 1.0f : 0.0f;
        }
    }
}

extern "C" void kernel_launch(void* const* d_in, const int* in_sizes, int n_in,
                              void* d_out, int out_size, void* d_ws, size_t ws_size,
                              hipStream_t stream) {
    (void)n_in; (void)out_size; (void)d_ws; (void)ws_size;
    const float* sh  = (const float*)d_in[0];
    const float* nbr = (const float*)d_in[1];
    const float* M   = (const float*)d_in[2];
    const float* W   = (const float*)d_in[3];
    const float* v0  = (const float*)d_in[4];
    const int*   pf  = (const int*)d_in[5];
    const int*   tf  = (const int*)d_in[6];
    const int*   tk  = (const int*)d_in[7];

    const int Btot = in_sizes[0] / 32;
    float* out_acc  = (float*)d_out;
    float* out_stay = (float*)d_out + (size_t)Btot * NUM_E;

    const int grid = (Btot + ROWS_PER_BLOCK - 1) / ROWS_PER_BLOCK;
    AgentLSM_cpg_kernel<<<grid, WAVES_PER_BLOCK * 32, 0, stream>>>(
        sh, nbr, M, W, v0, pf, tf, tk, out_acc, out_stay, Btot);
}
